// HybridAttention_29772713296033
// MI455X (gfx1250) — matrix operand order repaired and checked by reading, not compiled, not run on hardware
//
#include <hip/hip_runtime.h>
#include <hip/hip_bf16.h>

// ---------------------------------------------------------------------------
// HybridAttention for MI455X (gfx1250, wave32).
// Compute-bound (~115 GFLOP vs ~45MB data): all GEMMs via
// v_wmma_f32_16x16x32_bf16, bf16 operands / f32 accumulate.
// Weights pre-transposed to [N x K] bf16 so WMMA B-fragments are two
// contiguous 16B loads per lane; activations row-major [M x K] bf16 so
// A-fragments are two contiguous 16B loads per lane.
// ---------------------------------------------------------------------------

typedef __attribute__((ext_vector_type(16))) __bf16        v16bf;
typedef __attribute__((ext_vector_type(8)))  float         v8f;
typedef __attribute__((ext_vector_type(4)))  unsigned int  v4u;

#define D_MODEL 1024
#define SEQ     2048
#define BATCH   2
#define MTOK    (BATCH*SEQ)   /* 4096 */
#define NHEADS  16
#define HDIM    64
#define HALFW   32
#define NWIN    65

__device__ __forceinline__ unsigned short f2bf(float x) {
  unsigned int u = __float_as_uint(x);
  u += 0x7FFFu + ((u >> 16) & 1u);
  return (unsigned short)(u >> 16);
}

// ---------------------------------------------------------------------------
// Generic bf16 WMMA GEMM:  Y[M,N] = epilogue( A[M,K] @ Wt[N,K]^T + bias )
// Block: 256 thr = 8 waves in 2(M) x 4(N); wave tile 32x64; block tile 64x256.
// ---------------------------------------------------------------------------
enum { EPI_NONE = 0, EPI_ELU1 = 1, EPI_RELU = 2, EPI_GATECOMB = 3, EPI_RESID = 4 };

template <int EPI>
__global__ __launch_bounds__(256) void gemm_bf16_wmma(
    const unsigned short* __restrict__ A, int lda,
    const unsigned short* __restrict__ Wt,          // [N x K] bf16 (pre-transposed)
    const float* __restrict__ bias,                 // [N] or null
    const float* __restrict__ aux0,                 // gate-combine: lin ; resid: x
    const float* __restrict__ aux1,                 // gate-combine: win
    float* __restrict__ outF,                       // f32 out or null
    unsigned short* __restrict__ outBf,             // bf16 out or null
    int M, int N, int K)
{
  const int lane = threadIdx.x & 31;
  const int wave = threadIdx.x >> 5;
  const int wm = wave >> 2;              // 0..1
  const int wn = wave & 3;               // 0..3
  const int l16 = lane & 15;
  const int hi  = lane >> 4;             // half-wave select

  const int m_base = blockIdx.y * 64 + wm * 32;
  const int n_base = blockIdx.x * 256 + wn * 64;

  v8f acc[2][4] = {};
  union Frag { v4u u[2]; v16bf b; };

  for (int k0 = 0; k0 < K; k0 += 32) {
    Frag fa[2];
#pragma unroll
    for (int t = 0; t < 2; ++t) {
      // 16-bit A layout: lanes 0-15 hold K {0..7,16..23}, lanes 16-31 {8..15,24..31}
      const int row = m_base + t * 16 + l16;
      const unsigned short* p = A + (size_t)row * lda + k0 + hi * 8;
      fa[t].u[0] = *(const v4u*)(p);        // 8 bf16: K base..base+7
      fa[t].u[1] = *(const v4u*)(p + 16);   // 8 bf16: K base+16..base+23
    }
    Frag fb[4];
#pragma unroll
    for (int j = 0; j < 4; ++j) {
      // 16-bit B layout: lane = n; lanes 0-15 K {0..7,16..23}, lanes 16-31 {8..15,24..31}
      const int n = n_base + j * 16 + l16;
      const unsigned short* p = Wt + (size_t)n * K + k0 + hi * 8;
      fb[j].u[0] = *(const v4u*)(p);        // K base..base+7
      fb[j].u[1] = *(const v4u*)(p + 16);   // K base+16..base+23
    }
#pragma unroll
    for (int t = 0; t < 2; ++t)
#pragma unroll
      for (int j = 0; j < 4; ++j)
        acc[t][j] = __builtin_amdgcn_wmma_f32_16x16x32_bf16(
            false, fa[t].b, false, fb[j].b, (short)0, acc[t][j], false, false);
  }

  // C/D layout: lane n = l16; VGPR r -> row m = r + 8*hi within 16x16 tile
#pragma unroll
  for (int t = 0; t < 2; ++t) {
    const int m0 = m_base + t * 16 + hi * 8;
#pragma unroll
    for (int j = 0; j < 4; ++j) {
      const int n = n_base + j * 16 + l16;
      const float bval = bias ? bias[n] : 0.0f;
#pragma unroll
      for (int r = 0; r < 8; ++r) {
        const int m = m0 + r;
        const size_t idx = (size_t)m * N + n;
        float v = acc[t][j][r] + bval;
        if (EPI == EPI_ELU1) {
          v = (v > 0.0f) ? (v + 1.0f) : __expf(v);     // elu(v)+1
        } else if (EPI == EPI_RELU) {
          v = fmaxf(v, 0.0f);
        } else if (EPI == EPI_GATECOMB) {
          const float g = 1.0f / (1.0f + __expf(-v));  // sigmoid
          v = g * aux0[idx] + (1.0f - g) * aux1[idx];
        } else if (EPI == EPI_RESID) {
          v += aux0[idx];
        }
        if (outF)  outF[idx]  = v;
        if (outBf) outBf[idx] = f2bf(v);
      }
    }
  }
}

// ---------------------------------------------------------------------------
// LayerNorm: one block per token row; writes h as bf16 (standalone + concat slot 2)
// ---------------------------------------------------------------------------
__global__ __launch_bounds__(256) void layernorm_kernel(
    const float* __restrict__ x, const float* __restrict__ g,
    const float* __restrict__ b, unsigned short* __restrict__ h_bf,
    unsigned short* __restrict__ cat_bf)
{
  __shared__ float red[256];
  const int row = blockIdx.x;
  const float* xr = x + (size_t)row * D_MODEL;
  float vals[4];
  float s = 0.f;
#pragma unroll
  for (int i = 0; i < 4; ++i) { vals[i] = xr[threadIdx.x + 256 * i]; s += vals[i]; }
  red[threadIdx.x] = s; __syncthreads();
  for (int o = 128; o > 0; o >>= 1) { if (threadIdx.x < o) red[threadIdx.x] += red[threadIdx.x + o]; __syncthreads(); }
  const float mean = red[0] * (1.0f / D_MODEL);
  __syncthreads();
  float sq = 0.f;
#pragma unroll
  for (int i = 0; i < 4; ++i) { const float d = vals[i] - mean; sq += d * d; }
  red[threadIdx.x] = sq; __syncthreads();
  for (int o = 128; o > 0; o >>= 1) { if (threadIdx.x < o) red[threadIdx.x] += red[threadIdx.x + o]; __syncthreads(); }
  const float rstd = rsqrtf(red[0] * (1.0f / D_MODEL) + 1e-5f);
#pragma unroll
  for (int i = 0; i < 4; ++i) {
    const int c = threadIdx.x + 256 * i;
    const float hv = (vals[i] - mean) * rstd * g[c] + b[c];
    const unsigned short bf = f2bf(hv);
    h_bf[(size_t)row * D_MODEL + c] = bf;
    cat_bf[(size_t)row * (3 * D_MODEL) + 2 * D_MODEL + c] = bf;
  }
}

// ---------------------------------------------------------------------------
// Tiled transpose f32 [R,C] -> bf16 [C,R]  (weights -> [N,K]; k,v -> [D,S]; kv -> kv^T)
// ---------------------------------------------------------------------------
__global__ __launch_bounds__(256) void transpose_f32_bf16(
    const float* __restrict__ src, int R, int C, unsigned short* __restrict__ dst)
{
  __shared__ float tile[32][33];
  const int c0 = blockIdx.x * 32, r0 = blockIdx.y * 32;
  for (int i = threadIdx.y; i < 32; i += 8)
    tile[i][threadIdx.x] = src[(size_t)(r0 + i) * C + c0 + threadIdx.x];
  __syncthreads();
  for (int i = threadIdx.y; i < 32; i += 8)
    dst[(size_t)(c0 + i) * R + r0 + threadIdx.x] = f2bf(tile[threadIdx.x][i]);
}

// ---------------------------------------------------------------------------
// ksum[b][d] = sum_s k[b,s,d]
// ---------------------------------------------------------------------------
__global__ __launch_bounds__(256) void colsum_kernel(
    const float* __restrict__ k, float* __restrict__ ksum)
{
  const int idx = blockIdx.x * blockDim.x + threadIdx.x;  // 0..BATCH*D-1
  const int b = idx / D_MODEL, d = idx % D_MODEL;
  const float* p = k + (size_t)b * SEQ * D_MODEL + d;
  float s = 0.f;
  for (int si = 0; si < SEQ; ++si) s += p[(size_t)si * D_MODEL];
  ksum[idx] = s;
}

// ---------------------------------------------------------------------------
// lin_out = num / (q . ksum + 1e-6); in-place over num + bf16 into concat slot 0
// ---------------------------------------------------------------------------
__global__ __launch_bounds__(256) void lin_finalize_kernel(
    const float* __restrict__ qf, const float* __restrict__ ksum,
    float* __restrict__ num, unsigned short* __restrict__ cat_bf)
{
  __shared__ float red[256];
  const int row = blockIdx.x;           // 0..MTOK-1
  const int b = row / SEQ;
  const float* qr = qf + (size_t)row * D_MODEL;
  const float* ks = ksum + b * D_MODEL;
  float s = 0.f;
#pragma unroll
  for (int i = 0; i < 4; ++i) { const int c = threadIdx.x + 256 * i; s += qr[c] * ks[c]; }
  red[threadIdx.x] = s; __syncthreads();
  for (int o = 128; o > 0; o >>= 1) { if (threadIdx.x < o) red[threadIdx.x] += red[threadIdx.x + o]; __syncthreads(); }
  const float rden = 1.0f / (red[0] + 1e-6f);
#pragma unroll
  for (int i = 0; i < 4; ++i) {
    const int c = threadIdx.x + 256 * i;
    const size_t idx = (size_t)row * D_MODEL + c;
    const float v = num[idx] * rden;
    num[idx] = v;
    cat_bf[(size_t)row * (3 * D_MODEL) + c] = f2bf(v);
  }
}

// ---------------------------------------------------------------------------
// Sliding-window attention, flash-style online softmax.
// One wave32 per (token, head): lane owns 2 of the 64 head dims; scores via
// shfl_xor butterfly reduction. ~2 GFLOP total -> VALU is fine.
// ---------------------------------------------------------------------------
__global__ __launch_bounds__(256) void window_attn_kernel(
    const float* __restrict__ q, const float* __restrict__ k,
    const float* __restrict__ v, const float* __restrict__ rb,
    float* __restrict__ win, unsigned short* __restrict__ cat_bf)
{
  const int lane = threadIdx.x & 31;
  const int wave = threadIdx.x >> 5;
  const int gw = blockIdx.x * 8 + wave;       // 0 .. MTOK*NHEADS-1
  const int token = gw >> 4;                  // b*SEQ + s
  const int head  = gw & 15;
  const int b = token / SEQ;
  const int s = token % SEQ;

  const size_t qofs = (size_t)token * D_MODEL + head * HDIM + lane * 2;
  const float2 qv = *(const float2*)(q + qofs);
  const float scale = 0.125f;                 // 1/sqrt(64)

  float m = -3.0e38f, l = 0.f;
  float2 acc = make_float2(0.f, 0.f);
  for (int w = 0; w < NWIN; ++w) {
    const int s2 = s + w - HALFW;
    if (s2 < 0 || s2 >= SEQ) continue;
    const size_t kofs = ((size_t)(b * SEQ + s2)) * D_MODEL + head * HDIM + lane * 2;
    const float2 kk = *(const float2*)(k + kofs);
    float p = qv.x * kk.x + qv.y * kk.y;
#pragma unroll
    for (int o = 16; o > 0; o >>= 1) p += __shfl_xor(p, o, 32);
    const float sc = p * scale + rb[(size_t)(w + HALFW) * NHEADS + head];  // rel_bias[off+WINDOW]
    const float mnew = fmaxf(m, sc);
    const float corr = __expf(m - mnew);
    const float e    = __expf(sc - mnew);
    const float2 vv = *(const float2*)(v + kofs);
    acc.x = acc.x * corr + e * vv.x;
    acc.y = acc.y * corr + e * vv.y;
    l = l * corr + e;
    m = mnew;
  }
  const float rl = 1.0f / l;
  const float ox = acc.x * rl, oy = acc.y * rl;
  const size_t oofs = (size_t)token * D_MODEL + head * HDIM + lane * 2;
  win[oofs] = ox; win[oofs + 1] = oy;
  const size_t cofs = (size_t)token * (3 * D_MODEL) + D_MODEL + head * HDIM + lane * 2;
  cat_bf[cofs] = f2bf(ox); cat_bf[cofs + 1] = f2bf(oy);
}

// ---------------------------------------------------------------------------
// Host side
// ---------------------------------------------------------------------------
static inline void* ws_take(char*& cur, size_t bytes) {
  void* p = (void*)cur;
  cur += (bytes + 255) & ~(size_t)255;
  return p;
}

extern "C" void kernel_launch(void* const* d_in, const int* in_sizes, int n_in,
                              void* d_out, int out_size, void* d_ws, size_t ws_size,
                              hipStream_t stream) {
  (void)in_sizes; (void)n_in; (void)out_size; (void)ws_size;
  const float* x      = (const float*)d_in[0];
  const float* wq_lin = (const float*)d_in[1];  const float* bq_lin = (const float*)d_in[2];
  const float* wk_lin = (const float*)d_in[3];  const float* bk_lin = (const float*)d_in[4];
  const float* wv_lin = (const float*)d_in[5];  const float* bv_lin = (const float*)d_in[6];
  const float* wq_loc = (const float*)d_in[7];  const float* bq_loc = (const float*)d_in[8];
  const float* wk_loc = (const float*)d_in[9];  const float* bk_loc = (const float*)d_in[10];
  const float* wv_loc = (const float*)d_in[11]; const float* bv_loc = (const float*)d_in[12];
  const float* wo     = (const float*)d_in[13]; const float* bo     = (const float*)d_in[14];
  const float* wg1    = (const float*)d_in[15]; const float* bg1    = (const float*)d_in[16];
  const float* wg2    = (const float*)d_in[17]; const float* bg2    = (const float*)d_in[18];
  const float* rel_b  = (const float*)d_in[19];
  const float* ln_g   = (const float*)d_in[20]; const float* ln_b   = (const float*)d_in[21];
  float* out = (float*)d_out;

  const size_t TD  = (size_t)MTOK * D_MODEL;       // 4096*1024
  const size_t DD  = (size_t)D_MODEL * D_MODEL;
  const size_t SD  = (size_t)SEQ * D_MODEL;

  char* cur = (char*)d_ws;
  unsigned short* h_bf    = (unsigned short*)ws_take(cur, TD * 2);
  unsigned short* cat_bf  = (unsigned short*)ws_take(cur, TD * 3 * 2);
  unsigned short* wqlinT  = (unsigned short*)ws_take(cur, DD * 2);
  unsigned short* wklinT  = (unsigned short*)ws_take(cur, DD * 2);
  unsigned short* wvlinT  = (unsigned short*)ws_take(cur, DD * 2);
  unsigned short* wqlocT  = (unsigned short*)ws_take(cur, DD * 2);
  unsigned short* wklocT  = (unsigned short*)ws_take(cur, DD * 2);
  unsigned short* wvlocT  = (unsigned short*)ws_take(cur, DD * 2);
  unsigned short* woT     = (unsigned short*)ws_take(cur, DD * 2);
  unsigned short* wg2T    = (unsigned short*)ws_take(cur, DD * 2);
  unsigned short* wg1T    = (unsigned short*)ws_take(cur, DD * 3 * 2);
  float*          q_lin_f = (float*)ws_take(cur, TD * 4);
  unsigned short* q_lin_b = (unsigned short*)ws_take(cur, TD * 2);
  float*          k_lin_f = (float*)ws_take(cur, TD * 4);
  float*          v_lin_f = (float*)ws_take(cur, TD * 4);
  unsigned short* kT_bf   = (unsigned short*)ws_take(cur, (size_t)BATCH * SD * 2);
  unsigned short* vT_bf   = (unsigned short*)ws_take(cur, (size_t)BATCH * SD * 2);
  float*          kv_f    = (float*)ws_take(cur, (size_t)BATCH * DD * 4);
  unsigned short* kvT_bf  = (unsigned short*)ws_take(cur, (size_t)BATCH * DD * 2);
  float*          ksum    = (float*)ws_take(cur, (size_t)BATCH * D_MODEL * 4);
  float*          num_f   = (float*)ws_take(cur, TD * 4);   // becomes lin_out in place
  float*          q_loc_f = (float*)ws_take(cur, TD * 4);
  float*          k_loc_f = (float*)ws_take(cur, TD * 4);
  float*          v_loc_f = (float*)ws_take(cur, TD * 4);
  float*          win_f   = (float*)ws_take(cur, TD * 4);
  unsigned short* g1_bf   = (unsigned short*)ws_take(cur, TD * 2);
  unsigned short* y_bf    = (unsigned short*)ws_take(cur, TD * 2);

  const dim3 tb(32, 8);
  const dim3 gb(256);

  // 1. LayerNorm -> h (bf16)
  layernorm_kernel<<<MTOK, 256, 0, stream>>>(x, ln_g, ln_b, h_bf, cat_bf);

  // 2. Weight transposes [K,N] f32 -> [N,K] bf16
  transpose_f32_bf16<<<dim3(32, 32), tb, 0, stream>>>(wq_lin, D_MODEL, D_MODEL, wqlinT);
  transpose_f32_bf16<<<dim3(32, 32), tb, 0, stream>>>(wk_lin, D_MODEL, D_MODEL, wklinT);
  transpose_f32_bf16<<<dim3(32, 32), tb, 0, stream>>>(wv_lin, D_MODEL, D_MODEL, wvlinT);
  transpose_f32_bf16<<<dim3(32, 32), tb, 0, stream>>>(wq_loc, D_MODEL, D_MODEL, wqlocT);
  transpose_f32_bf16<<<dim3(32, 32), tb, 0, stream>>>(wk_loc, D_MODEL, D_MODEL, wklocT);
  transpose_f32_bf16<<<dim3(32, 32), tb, 0, stream>>>(wv_loc, D_MODEL, D_MODEL, wvlocT);
  transpose_f32_bf16<<<dim3(32, 32), tb, 0, stream>>>(wo,     D_MODEL, D_MODEL, woT);
  transpose_f32_bf16<<<dim3(32, 32), tb, 0, stream>>>(wg2,    D_MODEL, D_MODEL, wg2T);
  transpose_f32_bf16<<<dim3(32, 96), tb, 0, stream>>>(wg1, 3 * D_MODEL, D_MODEL, wg1T);

  // 3. Six projections (M=4096, N=1024, K=1024); elu+1 fused for linear branch q,k
  const dim3 pg(D_MODEL / 256, MTOK / 64);
  gemm_bf16_wmma<EPI_ELU1><<<pg, gb, 0, stream>>>(h_bf, D_MODEL, wqlinT, bq_lin, nullptr, nullptr, q_lin_f, q_lin_b, MTOK, D_MODEL, D_MODEL);
  gemm_bf16_wmma<EPI_ELU1><<<pg, gb, 0, stream>>>(h_bf, D_MODEL, wklinT, bk_lin, nullptr, nullptr, k_lin_f, nullptr, MTOK, D_MODEL, D_MODEL);
  gemm_bf16_wmma<EPI_NONE><<<pg, gb, 0, stream>>>(h_bf, D_MODEL, wvlinT, bv_lin, nullptr, nullptr, v_lin_f, nullptr, MTOK, D_MODEL, D_MODEL);
  gemm_bf16_wmma<EPI_NONE><<<pg, gb, 0, stream>>>(h_bf, D_MODEL, wqlocT, bq_loc, nullptr, nullptr, q_loc_f, nullptr, MTOK, D_MODEL, D_MODEL);
  gemm_bf16_wmma<EPI_NONE><<<pg, gb, 0, stream>>>(h_bf, D_MODEL, wklocT, bk_loc, nullptr, nullptr, k_loc_f, nullptr, MTOK, D_MODEL, D_MODEL);
  gemm_bf16_wmma<EPI_NONE><<<pg, gb, 0, stream>>>(h_bf, D_MODEL, wvlocT, bv_loc, nullptr, nullptr, v_loc_f, nullptr, MTOK, D_MODEL, D_MODEL);

  // 4. Linear attention: per-batch transposes, ksum, kv = k^T v, num = q @ kv
  for (int b = 0; b < BATCH; ++b) {
    transpose_f32_bf16<<<dim3(D_MODEL / 32, SEQ / 32), tb, 0, stream>>>(
        k_lin_f + (size_t)b * SD, SEQ, D_MODEL, kT_bf + (size_t)b * SD);
    transpose_f32_bf16<<<dim3(D_MODEL / 32, SEQ / 32), tb, 0, stream>>>(
        v_lin_f + (size_t)b * SD, SEQ, D_MODEL, vT_bf + (size_t)b * SD);
  }
  colsum_kernel<<<(BATCH * D_MODEL) / 256, 256, 0, stream>>>(k_lin_f, ksum);
  for (int b = 0; b < BATCH; ++b) {
    // kv[D,D] = kT[D,S] @ (vT[D,S])^T ; K = SEQ = 2048
    gemm_bf16_wmma<EPI_NONE><<<dim3(D_MODEL / 256, D_MODEL / 64), gb, 0, stream>>>(
        kT_bf + (size_t)b * SD, SEQ, vT_bf + (size_t)b * SD,
        nullptr, nullptr, nullptr, kv_f + (size_t)b * DD, nullptr,
        D_MODEL, D_MODEL, SEQ);
    transpose_f32_bf16<<<dim3(32, 32), tb, 0, stream>>>(
        kv_f + (size_t)b * DD, D_MODEL, D_MODEL, kvT_bf + (size_t)b * DD);
    // num[S,D] = q[S,D] @ kv[D,D]
    gemm_bf16_wmma<EPI_NONE><<<dim3(D_MODEL / 256, SEQ / 64), gb, 0, stream>>>(
        q_lin_b + (size_t)b * SD, D_MODEL, kvT_bf + (size_t)b * DD,
        nullptr, nullptr, nullptr, num_f + (size_t)b * SD, nullptr,
        SEQ, D_MODEL, D_MODEL);
  }
  lin_finalize_kernel<<<MTOK, 256, 0, stream>>>(q_lin_f, ksum, num_f, cat_bf);

  // 5. Window attention (flash-style), fills win_f + concat slot 1
  window_attn_kernel<<<(MTOK * NHEADS) / 8, 256, 0, stream>>>(
      q_loc_f, k_loc_f, v_loc_f, rel_b, win_f, cat_bf);

  // 6. Gate MLP: g1 = relu(cat @ wg1 + bg1)  (K = 3072)
  gemm_bf16_wmma<EPI_RELU><<<pg, gb, 0, stream>>>(
      cat_bf, 3 * D_MODEL, wg1T, bg1, nullptr, nullptr,
      nullptr, g1_bf, MTOK, D_MODEL, 3 * D_MODEL);

  // 7. gate = sigmoid(g1 @ wg2 + bg2); y = gate*lin + (1-gate)*win  (fused)
  gemm_bf16_wmma<EPI_GATECOMB><<<pg, gb, 0, stream>>>(
      g1_bf, D_MODEL, wg2T, bg2, num_f, win_f,
      nullptr, y_bf, MTOK, D_MODEL, D_MODEL);

  // 8. out = y @ wo + bo + x  (fused residual)
  gemm_bf16_wmma<EPI_RESID><<<pg, gb, 0, stream>>>(
      y_bf, D_MODEL, woT, bo, x, nullptr,
      out, nullptr, MTOK, D_MODEL, D_MODEL);
}
